// NetVladCNN_43258910605959
// MI455X (gfx1250) — compile-verified, matrix operand order checked by reading, not run on hardware
//
#include <hip/hip_runtime.h>
#include <hip/hip_bf16.h>

// NetVLAD: conv1x1 -> softmax(H) -> VLAD aggregation -> transpose + double L2-norm
// B=64, D=512, H=W=32 (N=1024), K=64.
//
// Memory-bound (~300MB @ 23.3 TB/s ~= 13us floor; 8.6 GFLOP of WMMA is free).
// conv GEMM: f32 x staged via regs (f32->f16 cvt), emits f16 copy xh; minimal-wait
//            barriers (s_wait_dscnt only) so next-tile global loads stay in flight.
// vlad GEMM: both tiles are pure f16 copies -> global_load_async_to_lds_b128,
//            synchronized with s_wait_asynccnt + split barrier.

#define B_  64
#define D_  512
#define N_  1024
#define K_  64
#define LSTR   40   // LDS row stride in halves (32 data + 8 pad) -> 80B rows, 16B aligned
#define LSTR32 20   // same stride in dwords

typedef __attribute__((ext_vector_type(16))) _Float16 v16h;
typedef __attribute__((ext_vector_type(8)))  float    v8f;

union Frag16 {
    v16h h;
    unsigned int u[8];
};

// ISA 7.12.2: 16-bit A(16x32)/B(32x16) fragment: VGPR j of this lane holds the
// K-pair kk = (j<4?0:16) + (lane>=16?8:0) + 2*(j&3). Return kk/2 (dword index).
__device__ __forceinline__ int kk_div2(int j, int lane) {
    return ((j < 4) ? 0 : 8) + ((lane & 16) ? 4 : 0) + (j & 3);
}

// Workgroup barrier that only drains LDS ops (leaves global loads/stores in flight).
__device__ __forceinline__ void wg_barrier_ds() {
    asm volatile("s_wait_dscnt 0x0\n\t"
                 "s_barrier_signal -1\n\t"
                 "s_barrier_wait -1" ::: "memory");
}

// Workgroup barrier that drains this wave's async global->LDS copies only.
__device__ __forceinline__ void wg_barrier_async() {
#if __has_builtin(__builtin_amdgcn_s_wait_asynccnt)
    __builtin_amdgcn_s_wait_asynccnt(0);
    asm volatile("s_barrier_signal -1\n\t"
                 "s_barrier_wait -1" ::: "memory");
#else
    asm volatile("s_wait_asynccnt 0x0\n\t"
                 "s_barrier_signal -1\n\t"
                 "s_barrier_wait -1" ::: "memory");
#endif
}

__device__ __forceinline__ void async_ld_b128(unsigned lds_byte_off, const void* gptr) {
    // per-lane 16B copy global -> LDS, tracked by ASYNCcnt
    asm volatile("global_load_async_to_lds_b128 %0, %1, off"
                 :: "v"(lds_byte_off), "v"(gptr) : "memory");
}

union HalfQuad { _Float16 h[4]; unsigned long long u64; };

// ---------------------------------------------------------------------------
// Kernel 0: zero asum accumulator
// ---------------------------------------------------------------------------
__global__ void zero_kernel(float* __restrict__ p, int n) {
    int t = blockIdx.x * blockDim.x + threadIdx.x;
    if (t < n) p[t] = 0.0f;
}

// ---------------------------------------------------------------------------
// Kernel 1: feat[b,k,n] = sum_d w[k,d]*x[b,d,n]; also emit xh = (f16)x.
// grid (N/64, B), block 128 (4 waves); wave m owns k-rows 16m..16m+15.
// Double-buffered LDS pipeline, one ds-only barrier per k-step.
// ---------------------------------------------------------------------------
__global__ __launch_bounds__(128)
void conv_gemm_kernel(const float* __restrict__ x, const float* __restrict__ w,
                      float* __restrict__ feat, _Float16* __restrict__ xh) {
    __shared__ __align__(16) _Float16 wt[2][K_ * LSTR];  // [k][d] 64x32
    __shared__ __align__(16) _Float16 xt[2][64 * LSTR];  // [n][d] 64x32 (transposed)

    const int n0   = blockIdx.x * 64;
    const int b    = blockIdx.y;
    const int t    = threadIdx.x;
    const int lane = t & 31;
    const int wave = t >> 5;

    const float* xb  = x  + (size_t)b * D_ * N_;
    _Float16*    xhb = xh + (size_t)b * D_ * N_;

    float4 wr[4], xr[4];

    auto load_regs = [&](int d0) {
        #pragma unroll
        for (int i = 0; i < 4; ++i) {          // w tile: 64k x 32d as float4
            int vi = t + i * 128;
            int k = vi >> 3, dq = (vi & 7) * 4;
            wr[i] = *(const float4*)&w[(size_t)k * D_ + d0 + dq];
        }
        #pragma unroll
        for (int i = 0; i < 4; ++i) {          // x tile: 32d x 64n as float4
            int vi = t + i * 128;
            int dl = vi >> 4, nq = (vi & 15) * 4;
            xr[i] = *(const float4*)&xb[(size_t)(d0 + dl) * N_ + n0 + nq];
        }
    };

    auto store_tiles = [&](int buf, int d0) {
        #pragma unroll
        for (int i = 0; i < 4; ++i) {
            int vi = t + i * 128;
            int k = vi >> 3, dq = (vi & 7) * 4;
            HalfQuad p;
            p.h[0] = (_Float16)wr[i].x; p.h[1] = (_Float16)wr[i].y;
            p.h[2] = (_Float16)wr[i].z; p.h[3] = (_Float16)wr[i].w;
            *(unsigned long long*)&wt[buf][k * LSTR + dq] = p.u64;   // ds_store_b64
        }
        #pragma unroll
        for (int i = 0; i < 4; ++i) {
            int vi = t + i * 128;
            int dl = vi >> 4, nq = (vi & 15) * 4;
            HalfQuad p;
            p.h[0] = (_Float16)xr[i].x; p.h[1] = (_Float16)xr[i].y;
            p.h[2] = (_Float16)xr[i].z; p.h[3] = (_Float16)xr[i].w;
            #pragma unroll
            for (int j = 0; j < 4; ++j)                              // transpose scatter
                xt[buf][(nq + j) * LSTR + dl] = p.h[j];
            *(unsigned long long*)&xhb[(size_t)(d0 + dl) * N_ + n0 + nq] = p.u64;
        }
    };

    v8f acc[4] = {};
    load_regs(0);

    for (int it = 0; it < D_ / 32; ++it) {
        const int cur = it & 1;
        store_tiles(cur, it * 32);                 // consumes regs of tile it
        if (it + 1 < D_ / 32) {
            load_regs((it + 1) * 32);              // issue next tile's loads now
            __builtin_prefetch(&xb[(size_t)(it * 32 + 64) * N_ + n0 + lane], 0, 1);
        }
        wg_barrier_ds();                           // drain LDS only; VMEM stays in flight

        const unsigned* wtU = (const unsigned*)&wt[cur][0];
        const unsigned* xtU = (const unsigned*)&xt[cur][0];
        Frag16 af;
        {
            int m = wave * 16 + (lane & 15);
            #pragma unroll
            for (int j = 0; j < 8; ++j) af.u[j] = wtU[m * LSTR32 + kk_div2(j, lane)];
        }
        #pragma unroll
        for (int nt = 0; nt < 4; ++nt) {
            Frag16 bf;
            int col = nt * 16 + (lane & 15);
            #pragma unroll
            for (int j = 0; j < 8; ++j) bf.u[j] = xtU[col * LSTR32 + kk_div2(j, lane)];
            acc[nt] = __builtin_amdgcn_wmma_f32_16x16x32_f16(
                false, af.h, false, bf.h, (short)0, acc[nt], false, false);
        }
    }

    #pragma unroll
    for (int nt = 0; nt < 4; ++nt) {
        int n = n0 + nt * 16 + (lane & 15);
        #pragma unroll
        for (int r = 0; r < 8; ++r) {
            int k = wave * 16 + r + ((lane & 16) ? 8 : 0);
            feat[((size_t)b * K_ + k) * N_ + n] = acc[nt][r];
        }
    }
}

// ---------------------------------------------------------------------------
// Kernel 2: softmax over H (32 values, stride W=32) per (b,k,w)
// ---------------------------------------------------------------------------
__global__ __launch_bounds__(256)
void softmax_kernel(const float* __restrict__ feat, _Float16* __restrict__ ah,
                    float* __restrict__ asum) {
    int t  = blockIdx.x * 256 + threadIdx.x;   // B*K*W threads
    int wq = t & 31;
    int k  = (t >> 5) & 63;
    int b  = t >> 11;
    size_t base = ((size_t)b * K_ + k) * N_ + wq;

    float v[32];
    float m = -1e30f;
    #pragma unroll
    for (int h = 0; h < 32; ++h) { v[h] = feat[base + (size_t)h * 32]; m = fmaxf(m, v[h]); }
    float s = 0.0f;
    #pragma unroll
    for (int h = 0; h < 32; ++h) { v[h] = __expf(v[h] - m); s += v[h]; }
    float inv = 1.0f / s;
    float part = 0.0f;
    #pragma unroll
    for (int h = 0; h < 32; ++h) {
        float e = v[h] * inv;
        part += e;
        ah[base + (size_t)h * 32] = (_Float16)e;
    }
    atomicAdd(&asum[b * K_ + k], part);
}

// ---------------------------------------------------------------------------
// Kernel 3: V[b,k,d] = sum_n a[b,k,n]*x[b,d,n] - asum[b,k]*c[k,d]
// Both tiles are f16 byte-copies -> staged with global_load_async_to_lds_b128
// (4 per thread per n-step), waited one step behind issue (ASYNCcnt).
// ---------------------------------------------------------------------------
__global__ __launch_bounds__(128)
void vlad_gemm_kernel(const _Float16* __restrict__ xh, const float* __restrict__ c,
                      const _Float16* __restrict__ ah, const float* __restrict__ asum,
                      float* __restrict__ V) {
    __shared__ __align__(16) _Float16 at[2][K_ * LSTR];  // [k][n] 64x32
    __shared__ __align__(16) _Float16 xt[2][64 * LSTR];  // [d][n] 64x32

    const int d0   = blockIdx.x * 64;
    const int b    = blockIdx.y;
    const int t    = threadIdx.x;
    const int lane = t & 31;
    const int wave = t >> 5;

    const _Float16* ab = ah + (size_t)b * K_ * N_;
    const _Float16* xb = xh + (size_t)b * D_ * N_ + (size_t)d0 * N_;

    const unsigned at_base  = (unsigned)(size_t)&at[0][0];
    const unsigned xt_base  = (unsigned)(size_t)&xt[0][0];
    const unsigned bufbytes = K_ * LSTR * 2;   // 5120, keeps 16B row alignment

    auto issue_async = [&](int buf, int n0) {
        #pragma unroll
        for (int i = 0; i < 2; ++i) {          // a tile: 64k x 32n = 256 x 16B
            int vi = t + i * 128;
            int k  = vi >> 2;
            int nq = (vi & 3) * 8;
            async_ld_b128(at_base + buf * bufbytes + (unsigned)(k * LSTR + nq) * 2,
                          (const void*)(ab + (size_t)k * N_ + n0 + nq));
        }
        #pragma unroll
        for (int i = 0; i < 2; ++i) {          // x tile: 64d x 32n = 256 x 16B
            int vi = t + i * 128;
            int dl = vi >> 2;
            int nq = (vi & 3) * 8;
            async_ld_b128(xt_base + buf * bufbytes + (unsigned)(dl * LSTR + nq) * 2,
                          (const void*)(xb + (size_t)dl * N_ + n0 + nq));
        }
    };

    v8f acc[4] = {};
    issue_async(0, 0);                         // prologue

    for (int it = 0; it < N_ / 32; ++it) {
        const int cur = it & 1;
        wg_barrier_async();                    // tile it landed + visible; prior reads done
        if (it + 1 < N_ / 32)
            issue_async(cur ^ 1, (it + 1) * 32);   // overlap with this step's WMMA

        const unsigned* atU = (const unsigned*)&at[cur][0];
        const unsigned* xtU = (const unsigned*)&xt[cur][0];
        Frag16 af;
        {
            int m = wave * 16 + (lane & 15);   // k row
            #pragma unroll
            for (int j = 0; j < 8; ++j) af.u[j] = atU[m * LSTR32 + kk_div2(j, lane)];
        }
        #pragma unroll
        for (int nt = 0; nt < 4; ++nt) {
            Frag16 bf;
            int col = nt * 16 + (lane & 15);   // d column
            #pragma unroll
            for (int j = 0; j < 8; ++j) bf.u[j] = xtU[col * LSTR32 + kk_div2(j, lane)];
            acc[nt] = __builtin_amdgcn_wmma_f32_16x16x32_f16(
                false, af.h, false, bf.h, (short)0, acc[nt], false, false);
        }
    }

    #pragma unroll
    for (int nt = 0; nt < 4; ++nt) {
        int d = d0 + nt * 16 + (lane & 15);
        #pragma unroll
        for (int r = 0; r < 8; ++r) {
            int k = wave * 16 + r + ((lane & 16) ? 8 : 0);
            float vv = acc[nt][r] - asum[b * K_ + k] * c[(size_t)k * D_ + d];
            V[((size_t)b * K_ + k) * D_ + d] = vv;
        }
    }
}

// ---------------------------------------------------------------------------
// Kernel 4: y = normalize(normalize(V.T(D,K,B), dim=K), dim=K)
// b fast-varying -> coalesced output stores; V reads hit L2 (just written, 8MB)
// ---------------------------------------------------------------------------
__global__ __launch_bounds__(256)
void normalize_kernel(const float* __restrict__ V, float* __restrict__ out) {
    int t = blockIdx.x * 256 + threadIdx.x;  // D*B threads
    int b = t & (B_ - 1);
    int d = t >> 6;
    float ss = 0.0f;
    for (int k = 0; k < K_; ++k) {
        float v = V[((size_t)b * K_ + k) * D_ + d];
        ss += v * v;
    }
    float n1 = fmaxf(sqrtf(ss), 1e-12f);
    float n2 = fmaxf(sqrtf(ss) / n1, 1e-12f);   // second normalize (~1.0)
    float scale = 1.0f / (n1 * n2);
    for (int k = 0; k < K_; ++k) {
        out[((size_t)d * K_ + k) * B_ + b] =
            V[((size_t)b * K_ + k) * D_ + d] * scale;
    }
}

// ---------------------------------------------------------------------------
// Launch
// ---------------------------------------------------------------------------
extern "C" void kernel_launch(void* const* d_in, const int* in_sizes, int n_in,
                              void* d_out, int out_size, void* d_ws, size_t ws_size,
                              hipStream_t stream) {
    (void)in_sizes; (void)n_in; (void)out_size; (void)ws_size;

    const float* x = (const float*)d_in[0];   // (B,D,H,W)
    const float* w = (const float*)d_in[1];   // (K,D)
    const float* c = (const float*)d_in[2];   // (K,D)
    float* out = (float*)d_out;               // (D,K,B)

    char* ws = (char*)d_ws;
    float*    feat = (float*)(ws);                    // 16 MiB (B,K,N) f32
    _Float16* ah   = (_Float16*)(ws + (16u << 20));   //  8 MiB (B,K,N) f16
    _Float16* xh   = (_Float16*)(ws + (24u << 20));   // 64 MiB (B,D,N) f16
    float*    asum = (float*)(ws + (88u << 20));      // 16 KiB (B,K)   f32
    float*    V    = (float*)(ws + (89u << 20));      //  8 MiB (B,K,D) f32

    zero_kernel<<<16, 256, 0, stream>>>(asum, B_ * K_);
    conv_gemm_kernel<<<dim3(N_ / 64, B_), 128, 0, stream>>>(x, w, feat, xh);
    softmax_kernel<<<(B_ * K_ * 32) / 256, 256, 0, stream>>>(feat, ah, asum);
    vlad_gemm_kernel<<<dim3(D_ / 64, B_), 128, 0, stream>>>(xh, c, ah, asum, V);
    normalize_kernel<<<(D_ * B_) / 256, 256, 0, stream>>>(V, out);
}
